// EOT_sample_54150947668631
// MI455X (gfx1250) — matrix operand order repaired and checked by reading, not compiled, not run on hardware
//
#include <hip/hip_runtime.h>

typedef __attribute__((ext_vector_type(16))) _Float16 v16h;
typedef __attribute__((ext_vector_type(8)))  float    v8f;

constexpr int   kNSrc  = 512;     // B (rows of C)
constexpr int   kNTgt  = 512;     // n samples (cols of C)
constexpr int   kLat   = 64;
constexpr int   kFcN   = 2048;    // 128*16
constexpr int   kPix   = 784;     // 28*28
constexpr int   kIter  = 1000;
constexpr float kReg   = 1.0f;

// ---------------------------------------------------------------------------
// Wave32 shuffle reductions + 2-barrier block reductions (LDS partials).
// ---------------------------------------------------------------------------
__device__ __forceinline__ float wredMax(float v) {
#pragma unroll
    for (int m = 16; m > 0; m >>= 1) v = fmaxf(v, __shfl_xor(v, m, 32));
    return v;
}
__device__ __forceinline__ float wredSum(float v) {
#pragma unroll
    for (int m = 16; m > 0; m >>= 1) v += __shfl_xor(v, m, 32);
    return v;
}
// All threads of the block must call; result returned to every thread.
__device__ __forceinline__ float blockMax(float v, float* wred) {
    int lane = threadIdx.x & 31, wid = threadIdx.x >> 5, nw = blockDim.x >> 5;
    v = wredMax(v);
    if (lane == 0) wred[wid] = v;
    __syncthreads();
    v = wredMax(lane < nw ? wred[lane] : -3.4e38f);
    __syncthreads();
    return v;
}
__device__ __forceinline__ float blockSum(float v, float* wred) {
    int lane = threadIdx.x & 31, wid = threadIdx.x >> 5, nw = blockDim.x >> 5;
    v = wredSum(v);
    if (lane == 0) wred[wid] = v;
    __syncthreads();
    v = wredSum(lane < nw ? wred[lane] : 0.0f);
    __syncthreads();
    return v;
}

// Pack 16 f32 (4x float4) into a WMMA f16 fragment.
__device__ __forceinline__ v16h cvt16(float4 a, float4 b, float4 c, float4 d) {
    v16h v;
    v[0]  = (_Float16)a.x; v[1]  = (_Float16)a.y; v[2]  = (_Float16)a.z; v[3]  = (_Float16)a.w;
    v[4]  = (_Float16)b.x; v[5]  = (_Float16)b.y; v[6]  = (_Float16)b.z; v[7]  = (_Float16)b.w;
    v[8]  = (_Float16)c.x; v[9]  = (_Float16)c.y; v[10] = (_Float16)c.z; v[11] = (_Float16)c.w;
    v[12] = (_Float16)d.x; v[13] = (_Float16)d.y; v[14] = (_Float16)d.z; v[15] = (_Float16)d.w;
    return v;
}

// ---------------------------------------------------------------------------
// FC: h = relu(z @ fc_w + fc_b); z[512,64] fc_w[64,2048] -> h f16 [512,2048]
// One 16x16 output tile per wave32, K=64 = 2 WMMA steps, float4 A loads.
// ---------------------------------------------------------------------------
__global__ __launch_bounds__(256)
void fc_wmma_kernel(const float* __restrict__ z, const float* __restrict__ w,
                    const float* __restrict__ bias, _Float16* __restrict__ h)
{
    const int M = kNSrc, K = kLat, N = kFcN;
    int wave = (blockIdx.x * blockDim.x + threadIdx.x) >> 5;
    int lane = threadIdx.x & 31;
    const int tilesN = N / 16;
    int tm = wave / tilesN;
    int tn = wave - tm * tilesN;
    if (tm >= M / 16) return;
    int g = lane >> 4;   // lane group (0: lanes 0-15, 1: lanes 16-31)
    int r = lane & 15;

    const float4* Arow = (const float4*)(z + (size_t)(tm * 16 + r) * K);
    v8f c = {};
#pragma unroll
    for (int kb = 0; kb < K; kb += 32) {
        // A (16x32 f16): lanes g=0 hold K=[kb..kb+7, kb+16..kb+23], g=1 +8
        int sA0 = (kb + g * 8) >> 2;
        int sA1 = (kb + 16 + g * 8) >> 2;
        v16h a = cvt16(Arow[sA0], Arow[sA0 + 1], Arow[sA1], Arow[sA1 + 1]);
        // B (32x16 f16): same K order as A: g=0 K=[kb..kb+7, kb+16..kb+23], g=1 +8
        v16h b;
#pragma unroll
        for (int e = 0; e < 16; ++e) {
            int kn = kb + (e & 7) + (e >> 3) * 16 + g * 8;
            b[e] = (_Float16)w[(size_t)kn * N + tn * 16 + r];
        }
        c = __builtin_amdgcn_wmma_f32_16x16x32_f16(false, a, false, b,
                                                   (short)0, c, false, false);
    }
#pragma unroll
    for (int r8 = 0; r8 < 8; ++r8) {
        int m = tm * 16 + g * 8 + r8;   // C/D layout: VGPR r8 -> M = r8 + g*8
        int n = tn * 16 + r;
        float v = c[r8] + bias[n];
        h[(size_t)m * N + n] = (_Float16)fmaxf(v, 0.0f);
    }
}

// ---------------------------------------------------------------------------
// ConvTranspose2d k=3 s=2 p=1 (+output_padding via Ho/Wo), w[Ci,Co,3,3], relu.
// ---------------------------------------------------------------------------
__global__ __launch_bounds__(256)
void convt_kernel(const _Float16* __restrict__ x, const float* __restrict__ w,
                  const float* __restrict__ bias, _Float16* __restrict__ y,
                  int Nb, int Ci, int Hi, int Wi, int Co, int Ho, int Wo)
{
    int t = blockIdx.x * blockDim.x + threadIdx.x;
    int total = Nb * Co * Ho * Wo;
    if (t >= total) return;
    int ox = t % Wo; int tmp = t / Wo;
    int oy = tmp % Ho; tmp /= Ho;
    int co = tmp % Co; int n = tmp / Co;

    float acc = bias[co];
    for (int ky = 0; ky < 3; ++ky) {
        int ty = oy + 1 - ky;
        if (ty & 1) continue;
        int iy = ty >> 1;
        if (iy < 0 || iy >= Hi) continue;
        for (int kx = 0; kx < 3; ++kx) {
            int tx = ox + 1 - kx;
            if (tx & 1) continue;
            int ix = tx >> 1;
            if (ix < 0 || ix >= Wi) continue;
            const _Float16* xp = x + (((size_t)n * Ci) * Hi + iy) * Wi + ix;
            const float*    wp = w + (size_t)co * 9 + ky * 3 + kx;
            for (int ci = 0; ci < Ci; ++ci)
                acc += (float)xp[(size_t)ci * Hi * Wi] * wp[(size_t)ci * Co * 9];
        }
    }
    y[t] = (_Float16)fmaxf(acc, 0.0f);
}

// ---------------------------------------------------------------------------
// Final Conv2d 32->1, k=3, p=1, OIHW weights, no activation -> f32 logits.
// ---------------------------------------------------------------------------
__global__ __launch_bounds__(256)
void conv_out_kernel(const _Float16* __restrict__ x, const float* __restrict__ w,
                     const float* __restrict__ bias, float* __restrict__ logits,
                     int Nb, int Ci, int H, int W)
{
    int t = blockIdx.x * blockDim.x + threadIdx.x;
    int total = Nb * H * W;
    if (t >= total) return;
    int ox = t % W;
    int oy = (t / W) % H;
    int n  = t / (W * H);

    float acc = bias[0];
    for (int ci = 0; ci < Ci; ++ci) {
        for (int ky = 0; ky < 3; ++ky) {
            int iy = oy - 1 + ky;
            if (iy < 0 || iy >= H) continue;
            for (int kx = 0; kx < 3; ++kx) {
                int ix = ox - 1 + kx;
                if (ix < 0 || ix >= W) continue;
                acc += (float)x[(((size_t)n * Ci + ci) * H + iy) * W + ix]
                       * w[(ci * 3 + ky) * 3 + kx];
            }
        }
    }
    logits[t] = acc;
}

// ---------------------------------------------------------------------------
// sp[j] = sum_p softplus(logits[j,p]); one 256-thread block per row.
// ---------------------------------------------------------------------------
__global__ __launch_bounds__(256)
void softplus_kernel(const float* __restrict__ logits, float* __restrict__ sp)
{
    __shared__ float wred[8];
    int j = blockIdx.x;
    float s = 0.0f;
    for (int p = threadIdx.x; p < kPix; p += 256) {
        float v = logits[(size_t)j * kPix + p];
        s += (v > 0.0f) ? (v + log1pf(expf(-v))) : log1pf(expf(v));
    }
    s = blockSum(s, wred);
    if (threadIdx.x == 0) sp[j] = s;
}

// ---------------------------------------------------------------------------
// Cost matrix: C[i,j] = sp[j] - dot(X[i,:], L[j,:]); WMMA GEMM, K=784 (pad 800).
// Per wave: 16x64 output tile -> A fragment reused across 4 WMMAs per K-step.
// All fragment loads are float4 (784 = 8*98 = 16*49, so tail guards are
// whole-chunk and loads stay vectorized).
// ---------------------------------------------------------------------------
__global__ __launch_bounds__(256)
void cost_wmma_kernel(const float* __restrict__ X, const float* __restrict__ L,
                      const float* __restrict__ sp, float* __restrict__ Cmat)
{
    const int M = kNSrc, N = kNTgt, K = kPix;
    int wave = (blockIdx.x * blockDim.x + threadIdx.x) >> 5;
    int lane = threadIdx.x & 31;
    const int tilesN64 = N / 64;   // 8
    int tm   = wave / tilesN64;
    int tn64 = wave - tm * tilesN64;
    if (tm >= M / 16) return;
    int g = lane >> 4;
    int r = lane & 15;

    const float4* Arow = (const float4*)(X + (size_t)(tm * 16 + r) * K);
    const float4* Brow[4];
#pragma unroll
    for (int s = 0; s < 4; ++s)
        Brow[s] = (const float4*)(L + (size_t)(tn64 * 64 + s * 16 + r) * K);

    v8f acc[4] = {v8f{}, v8f{}, v8f{}, v8f{}};
    const float4 z4 = {0.0f, 0.0f, 0.0f, 0.0f};

    for (int kb = 0; kb < 800; kb += 32) {   // 25 steps, last zero-padded
        // A fragment: 16 consecutive floats at kb+g*16 (two 8-float runs)
        int sA0 = kb + g * 16;
        int sA1 = kb + g * 16 + 8;
        float4 a0 = z4, a1 = z4, a2 = z4, a3 = z4;
        if (sA0 < K) { a0 = Arow[sA0 >> 2]; a1 = Arow[(sA0 >> 2) + 1]; }
        if (sA1 < K) { a2 = Arow[sA1 >> 2]; a3 = Arow[(sA1 >> 2) + 1]; }
        v16h a = cvt16(a0, a1, a2, a3);

        // B fragments: 16 consecutive floats at kb+g*16 of row n
        int sB = kb + g * 16;
        bool okB = sB < K;
#pragma unroll
        for (int s = 0; s < 4; ++s) {
            float4 b0 = z4, b1 = z4, b2 = z4, b3 = z4;
            if (okB) {
                const float4* bp = Brow[s] + (sB >> 2);
                b0 = bp[0]; b1 = bp[1]; b2 = bp[2]; b3 = bp[3];
            }
            v16h b = cvt16(b0, b1, b2, b3);
            acc[s] = __builtin_amdgcn_wmma_f32_16x16x32_f16(
                false, a, false, b, (short)0, acc[s], false, false);
        }
    }
#pragma unroll
    for (int s = 0; s < 4; ++s) {
#pragma unroll
        for (int r8 = 0; r8 < 8; ++r8) {
            int m = tm * 16 + g * 8 + r8;
            int n = tn64 * 64 + s * 16 + r;
            Cmat[(size_t)m * N + n] = sp[n] - acc[s][r8];
        }
    }
}

// ---------------------------------------------------------------------------
// Sequential SAG scan: single 512-thread block (16 waves), thread j owns
// column j. Softmax reductions via wave32 shuffles: 4 barriers/iteration.
// ---------------------------------------------------------------------------
__global__ __launch_bounds__(512)
void sag_kernel(const float* __restrict__ Cmat, const int* __restrict__ idx,
                float* __restrict__ stored, float* __restrict__ beta_out)
{
    __shared__ float wred[16];
    __shared__ int   sidx[kIter];
    int j = threadIdx.x;
    for (int t = j; t < kIter; t += kNTgt) sidx[t] = idx[t];
    for (int i = 0; i < kNSrc; ++i) stored[(size_t)i * kNTgt + j] = 0.0f;
    __syncthreads();

    const float a_i   = 1.0f / kNSrc;
    const float b_j   = 1.0f / kNTgt;
    const float log_b = logf(b_j);
    float beta = 0.0f, ssum = 0.0f;

    for (int t = 0; t < kIter; ++t) {
        int row = sidx[t];
        float u = (beta - Cmat[(size_t)row * kNTgt + j]) / kReg + log_b;
        float umax  = blockMax(u, wred);
        float e     = expf(u - umax);
        float denom = blockSum(e, wred);

        float khi  = e / denom;
        float gj   = a_i * (b_j - khi);
        float prev = stored[(size_t)row * kNTgt + j];
        ssum += gj - prev;
        stored[(size_t)row * kNTgt + j] = gj;
        beta += kReg * ssum;   // lr/n_src = reg
    }
    beta_out[j] = beta;
}

// ---------------------------------------------------------------------------
// alpha[i] = -reg * logsumexp_j((beta_j - C[i,j])/reg + log_b); block per row.
// ---------------------------------------------------------------------------
__global__ __launch_bounds__(512)
void alpha_kernel(const float* __restrict__ Cmat, const float* __restrict__ beta,
                  float* __restrict__ alpha)
{
    __shared__ float wred[16];
    int i = blockIdx.x;
    int j = threadIdx.x;
    const float log_b = logf(1.0f / kNTgt);
    float u = (beta[j] - Cmat[(size_t)i * kNTgt + j]) / kReg + log_b;
    float umax = blockMax(u, wred);
    float e    = expf(u - umax);
    float sum  = blockSum(e, wred);
    if (j == 0) alpha[i] = -kReg * (umax + logf(sum));
}

// ---------------------------------------------------------------------------
// P[i,j] = exp((alpha_i + beta_j - C_ij)/reg)*a*b; per-block partial of
// sum P*log(P) for a deterministic two-stage reduction.
// ---------------------------------------------------------------------------
__global__ __launch_bounds__(256)
void p_kl_kernel(const float* __restrict__ Cmat, const float* __restrict__ alpha,
                 const float* __restrict__ beta, float* __restrict__ P,
                 float* __restrict__ partials)
{
    __shared__ float wred[8];
    int t = blockIdx.x * 256 + threadIdx.x;
    int i = t / kNTgt;
    int j = t - i * kNTgt;
    float p = expf((alpha[i] + beta[j] - Cmat[t]) / kReg)
              * (1.0f / kNSrc) * (1.0f / kNTgt);
    P[t] = p;
    float xs = (p <= 0.0f) ? 1.0f : p;
    float s = blockSum(xs * logf(xs), wred);
    if (threadIdx.x == 0) partials[blockIdx.x] = s;
}

__global__ __launch_bounds__(256)
void kl_final_kernel(const float* __restrict__ partials, int n,
                     float* __restrict__ kl_out)
{
    __shared__ float wred[8];
    float s = 0.0f;
    for (int t = threadIdx.x; t < n; t += 256) s += partials[t];
    s = blockSum(s, wred);
    if (threadIdx.x == 0) {
        float omega_a = kNSrc * ((1.0f / kNSrc) * logf(1.0f / kNSrc));
        float omega_b = kNTgt * ((1.0f / kNTgt) * logf(1.0f / kNTgt));
        kl_out[0] = s - omega_a - omega_b;
    }
}

// ---------------------------------------------------------------------------
extern "C" void kernel_launch(void* const* d_in, const int* in_sizes, int n_in,
                              void* d_out, int out_size, void* d_ws, size_t ws_size,
                              hipStream_t stream)
{
    (void)in_sizes; (void)n_in; (void)out_size; (void)ws_size;

    const float* x     = (const float*)d_in[0];   // [512,1,28,28]
    const float* z     = (const float*)d_in[1];   // [512,64]
    const float* fc_w  = (const float*)d_in[2];   // [64,2048]
    const float* fc_b  = (const float*)d_in[3];
    const float* ct0_w = (const float*)d_in[4];   // [128,64,3,3]
    const float* ct0_b = (const float*)d_in[5];
    const float* ct1_w = (const float*)d_in[6];   // [64,32,3,3]
    const float* ct1_b = (const float*)d_in[7];
    const float* fct_w = (const float*)d_in[8];   // [32,32,3,3]
    const float* fct_b = (const float*)d_in[9];
    const float* cv_w  = (const float*)d_in[10];  // [1,32,3,3]
    const float* cv_b  = (const float*)d_in[11];
    const int*   idx   = (const int*)d_in[12];    // [1000]

    float* P_out  = (float*)d_out;
    float* C_out  = P_out + (size_t)kNSrc * kNTgt;
    float* kl_out = C_out + (size_t)kNSrc * kNTgt;

    char* wp = (char*)d_ws;
    auto alloc = [&](size_t bytes) -> char* {
        char* r = wp;
        wp += (bytes + 255) & ~(size_t)255;
        return r;
    };
    _Float16* h      = (_Float16*)alloc((size_t)512 * 2048 * sizeof(_Float16));
    _Float16* a0     = (_Float16*)alloc((size_t)512 * 64 * 7 * 7 * sizeof(_Float16));
    _Float16* a1     = (_Float16*)alloc((size_t)512 * 32 * 14 * 14 * sizeof(_Float16));
    _Float16* a2     = (_Float16*)alloc((size_t)512 * 32 * 28 * 28 * sizeof(_Float16));
    float* logits    = (float*)alloc((size_t)512 * kPix * sizeof(float));
    float* sp        = (float*)alloc(512 * sizeof(float));
    float* stored    = (float*)alloc((size_t)kNSrc * kNTgt * sizeof(float));
    float* beta      = (float*)alloc(512 * sizeof(float));
    float* alpha     = (float*)alloc(512 * sizeof(float));
    float* partials  = (float*)alloc(1024 * sizeof(float));

    // 1) FC + relu (WMMA): (512/16)*(2048/16)=4096 waves, 8 waves/block
    fc_wmma_kernel<<<512, 256, 0, stream>>>(z, fc_w, fc_b, h);

    // 2) ConvT 128->64, 4x4 -> 7x7
    {
        int total = 512 * 64 * 7 * 7;
        convt_kernel<<<(total + 255) / 256, 256, 0, stream>>>(
            h, ct0_w, ct0_b, a0, 512, 128, 4, 4, 64, 7, 7);
    }
    // 3) ConvT 64->32, 7x7 -> 14x14
    {
        int total = 512 * 32 * 14 * 14;
        convt_kernel<<<(total + 255) / 256, 256, 0, stream>>>(
            a0, ct1_w, ct1_b, a1, 512, 64, 7, 7, 32, 14, 14);
    }
    // 4) ConvT 32->32, 14x14 -> 28x28
    {
        int total = 512 * 32 * 28 * 28;
        convt_kernel<<<(total + 255) / 256, 256, 0, stream>>>(
            a1, fct_w, fct_b, a2, 512, 32, 14, 14, 32, 28, 28);
    }
    // 5) Conv 32->1 -> logits
    {
        int total = 512 * 28 * 28;
        conv_out_kernel<<<(total + 255) / 256, 256, 0, stream>>>(
            a2, cv_w, cv_b, logits, 512, 32, 28, 28);
    }
    // 6) softplus row sums
    softplus_kernel<<<512, 256, 0, stream>>>(logits, sp);

    // 7) Cost matrix (WMMA): (512/16)*(512/64)=256 waves, 8 waves/block
    cost_wmma_kernel<<<32, 256, 0, stream>>>(x, logits, sp, C_out);

    // 8) Sequential SAG scan (1000 iterations), single block
    sag_kernel<<<1, 512, 0, stream>>>(C_out, idx, stored, beta);

    // 9) entropic c-transform
    alpha_kernel<<<512, 512, 0, stream>>>(C_out, beta, alpha);

    // 10) primal plan + omega partials
    p_kl_kernel<<<1024, 256, 0, stream>>>(C_out, alpha, beta, P_out, partials);

    // 11) KL scalar
    kl_final_kernel<<<1, 256, 0, stream>>>(partials, 1024, kl_out);
}
